// ContactAwareStateTransitionAdapter_29283087024825
// MI455X (gfx1250) — matrix operand order repaired and checked by reading, not compiled, not run on hardware
//
#include <hip/hip_runtime.h>
#include <hip/hip_bf16.h>
#include <math.h>

// ---------------------------------------------------------------------------
// MI455X (gfx1250) implementation. All heavy GEMMs run on
// v_wmma_f32_16x16x32_f16 (wave32 WMMA). fp32 activations are converted to
// f16 in LDS tiles; fp32 weights are pre-converted once per launch (no
// transpose: reference weights are [out][in] == [N][K] row-major, which is
// exactly the n-major layout the LDS B tiles want).
// B tiles stream via GLOBAL_LOAD_ASYNC_TO_LDS_B128 (ASYNCcnt-tracked, no
// VGPR round-trip) when the toolchain exposes the builtin.
// ---------------------------------------------------------------------------

typedef __attribute__((ext_vector_type(16))) _Float16 v16h;
typedef __attribute__((ext_vector_type(8)))  _Float16 v8h;
typedef __attribute__((ext_vector_type(8)))  float    v8f;
typedef int v4i __attribute__((vector_size(16)));
typedef __attribute__((address_space(1))) v4i* gptr_v4i;
typedef __attribute__((address_space(3))) v4i* lptr_v4i;

#if defined(__HIP_DEVICE_COMPILE__) && __has_builtin(__builtin_amdgcn_global_load_async_to_lds_b128)
#define ASYNC_B 1
#else
#define ASYNC_B 0
#endif

#define DEV __device__ __forceinline__

DEV float gelu_f(float x) { return 0.5f * x * (1.0f + erff(x * 0.70710678118654752f)); }

// row remap: logical row r -> physical row (r/g)*(g+p) + h + r%g
DEV long rmap(int r, int g, int p, int h) {
  return (long)(r / g) * (long)(g + p) + (long)h + (long)(r % g);
}

DEV void wait_async_zero() {
#if ASYNC_B
#if __has_builtin(__builtin_amdgcn_s_wait_asynccnt)
  __builtin_amdgcn_s_wait_asynccnt(0);
#else
  asm volatile("s_wait_asynccnt 0x0" ::: "memory");
#endif
#endif
}

// ---------------------------------------------------------------------------
// Generic WMMA GEMM: C[M,N] = epilogue(A[M,K] @ W[N,K]^T + bias)
//   A: fp32, row-remapped (ag,apd,ah), leading dim lda
//   Bt: f16, N x K row-major (ldb == K)  [native weight layout]
//   C: fp32, row-remapped (cg,cp,ch), leading dim ldc
//   mode bit0: GELU; bit1: add residual (resid uses C remap, leading dim ldr)
// Block: 256 threads (8 waves), 128x64 tile, K step 32.
// Wave w owns rows [16w,16w+16), all four 16-col tiles -> 4 WMMA / k-step,
// A fragment reused 4x.
// ---------------------------------------------------------------------------
__global__ __launch_bounds__(256) void gemm_wmma_kernel(
    const float* __restrict__ A, int lda, int ag, int apd, int ah,
    const _Float16* __restrict__ Bt, int ldb,
    const float* __restrict__ bias,
    const float* __restrict__ resid, int ldr,
    float* __restrict__ C, int ldc, int cg, int cp, int ch,
    int M, int N, int K, int mode) {
  __shared__ __align__(16) _Float16 lA[128 * 32];  // [m][k]
  __shared__ __align__(16) _Float16 lB[64 * 32];   // [n][k]

  const int tid  = threadIdx.x;
  const int m0   = blockIdx.y * 128;
  const int n0   = blockIdx.x * 64;
  const int w    = tid >> 5;
  const int lane = tid & 31;
  const int mt   = w * 16;             // wave's 16-row stripe in tile
  const int g    = lane >> 4;          // lane group (0: lanes 0-15, 1: 16-31)
  const int l16  = lane & 15;

  v8f acc[4] = {v8f{}, v8f{}, v8f{}, v8f{}};

  // A-staging mapping: thread -> row tid>>1 (0..127), 16 cols at (tid&1)*16
  const int arow_ld = tid >> 1;
  const int acol_ld = (tid & 1) * 16;
  const long asrc_row = rmap(m0 + arow_ld, ag, apd, ah);
  // B-staging mapping: thread -> n = tid>>2 (0..63), 8 k at (tid&3)*8
  const int bn_ld = tid >> 2;
  const int bk_ld = (tid & 3) * 8;

  for (int k0 = 0; k0 < K; k0 += 32) {
    // --- stage A tile: 128 rows x 32 cols fp32 -> f16 LDS ---
    {
      const float* src = A + asrc_row * (long)lda + k0 + acol_ld;
      _Float16* dst = lA + arow_ld * 32 + acol_ld;
#pragma unroll
      for (int i = 0; i < 16; ++i) dst[i] = (_Float16)src[i];
    }
    // --- stage B tile: 64 n-rows x 32 k, contiguous 16B in global and LDS ---
    {
      const _Float16* src = Bt + (long)(n0 + bn_ld) * ldb + k0 + bk_ld;
      _Float16* dst = lB + bn_ld * 32 + bk_ld;
#if ASYNC_B
      __builtin_amdgcn_global_load_async_to_lds_b128((gptr_v4i)src, (lptr_v4i)dst, 0, 0);
#else
#pragma unroll
      for (int i = 0; i < 8; ++i) dst[i] = src[i];
#endif
    }
    wait_async_zero();
    __syncthreads();

    // --- fragments: A (2x ds_load_b128), B (2x ds_load_b128 per tile) ---
    union { v16h v; v8h h[2]; } ua;
    const v8h* arow = (const v8h*)(lA + (mt + l16) * 32);
    ua.h[0] = arow[g];        // K = 8g .. 8g+7
    ua.h[1] = arow[2 + g];    // K = 16+8g .. 16+8g+7
#pragma unroll
    for (int q = 0; q < 4; ++q) {
      union { v16h v; v8h h[2]; } ub;
      const v8h* brow = (const v8h*)(lB + (q * 16 + l16) * 32);
      ub.h[0] = brow[g];          // K = 8g .. 8g+7
      ub.h[1] = brow[2 + g];      // K = 16+8g .. 16+8g+7
      acc[q] = __builtin_amdgcn_wmma_f32_16x16x32_f16(false, ua.v, false, ub.v,
                                                      (short)0, acc[q], false, false);
    }
    __syncthreads();
  }

  // --- epilogue: D layout is M = r + 8*g (VGPR r), N = lane%16 ---
  long crow[8];
#pragma unroll
  for (int r = 0; r < 8; ++r) crow[r] = rmap(m0 + mt + r + 8 * g, cg, cp, ch);
#pragma unroll
  for (int q = 0; q < 4; ++q) {
    const int gn = n0 + q * 16 + l16;
    const float bv = bias ? bias[gn] : 0.0f;
#pragma unroll
    for (int r = 0; r < 8; ++r) {
      float v = acc[q][r] + bv;
      if (mode & 1) v = gelu_f(v);
      if (mode & 2) v += resid[crow[r] * (long)ldr + gn];
      C[crow[r] * (long)ldc + gn] = v;
    }
  }
}

// ---------------------------------------------------------------------------
// Weight convert: fp32 -> f16, same layout (weights are already [N][K])
// ---------------------------------------------------------------------------
__global__ void wcast_kernel(const float* __restrict__ W, _Float16* __restrict__ O, long n) {
  long i = (long)blockIdx.x * 256 + threadIdx.x;
  if (i < n) O[i] = (_Float16)W[i];
}

// ---------------------------------------------------------------------------
// delta + saliency: one wave per (b,t,n) row of 192
// ---------------------------------------------------------------------------
__global__ void delta_sal_kernel(const float* __restrict__ ph,
                                 float* __restrict__ delta,
                                 float* __restrict__ sal) {
  const int row  = blockIdx.x * 8 + (threadIdx.x >> 5);
  const int lane = threadIdx.x & 31;
  const int t = (row / 196) & 15;
  const float* cur  = ph + (long)row * 192;
  const float* prev = cur - (long)196 * 192;
  float asum = 0.0f;
  for (int i = lane; i < 192; i += 32) {
    float d = (t == 0) ? 0.0f : (cur[i] - prev[i]);
    delta[(long)row * 192 + i] = d;
    asum += fabsf(d);
  }
  for (int off = 16; off > 0; off >>= 1) asum += __shfl_xor(asum, off, 32);
  if (lane == 0) sal[row] = asum * (1.0f / 192.0f);
}

__global__ void dp_kernel(const float* __restrict__ delta, float* __restrict__ dp) {
  int i = blockIdx.x * blockDim.x + threadIdx.x;   // b*192 + a
  if (i >= 16 * 192) return;
  int b = i / 192, a = i % 192;
  float s = 0.0f;
  const float* base = delta + (long)b * 16 * 196 * 192 + a;
  for (int tn = 0; tn < 16 * 196; ++tn) s += fabsf(base[(long)tn * 192]);
  dp[i] = s * (1.0f / (16.0f * 196.0f));
}

// depthwise 3x3 SAME conv on (bt,14,14,a) + GELU
__global__ void conv3x3_kernel(const float* __restrict__ ph, const float* __restrict__ w,
                               float* __restrict__ out) {
  long i = (long)blockIdx.x * 256 + threadIdx.x;
  if (i >= (long)50176 * 192) return;
  int a = (int)(i % 192);
  long r = i / 192;
  int n = (int)(r % 196);
  long bt = r / 196;
  int y = n / 14, x = n % 14;
  const float* wp = w + a * 9;
  float acc = 0.0f;
#pragma unroll
  for (int dy = -1; dy <= 1; ++dy)
#pragma unroll
    for (int dx = -1; dx <= 1; ++dx) {
      int yy = y + dy, xx = x + dx;
      if (yy >= 0 && yy < 14 && xx >= 0 && xx < 14)
        acc += ph[((bt * 196) + yy * 14 + xx) * 192 + a] * wp[(dy + 1) * 3 + (dx + 1)];
    }
  out[i] = gelu_f(acc);
}

// depthwise temporal conv (kernel 3, SAME) on (b,t,n,a) + GELU
__global__ void tconv_kernel(const float* __restrict__ delta, const float* __restrict__ w,
                             float* __restrict__ out) {
  long i = (long)blockIdx.x * 256 + threadIdx.x;
  if (i >= (long)50176 * 192) return;
  int a = (int)(i % 192);
  long r = i / 192;
  long btq = r / 196;
  int t = (int)(btq % 16);
  const float* wp = w + a * 3;
  const long stride = (long)196 * 192;
  float acc = delta[i] * wp[1];
  if (t > 0)  acc += delta[i - stride] * wp[0];
  if (t < 15) acc += delta[i + stride] * wp[2];
  out[i] = gelu_f(acc);
}

// row LayerNorm over feat=192: one wave per row
__global__ void ln_kernel(const float* __restrict__ in, float* __restrict__ out,
                          const float* __restrict__ gam, const float* __restrict__ bet) {
  const int row  = blockIdx.x * 8 + (threadIdx.x >> 5);
  const int lane = threadIdx.x & 31;
  const float* x = in + (long)row * 192;
  float s = 0.0f, s2 = 0.0f;
  for (int i = lane; i < 192; i += 32) { float v = x[i]; s += v; s2 += v * v; }
  for (int off = 16; off > 0; off >>= 1) {
    s  += __shfl_xor(s,  off, 32);
    s2 += __shfl_xor(s2, off, 32);
  }
  const float m   = s * (1.0f / 192.0f);
  const float var = s2 * (1.0f / 192.0f) - m * m;
  const float inv = rsqrtf(var + 1e-5f);
  float* o = out + (long)row * 192;
  for (int i = lane; i < 192; i += 32) o[i] = (x[i] - m) * inv * gam[i] + bet[i];
}

// top-8 of sal[196] per (b,t): one block per (b,t)
__global__ void topk_kernel(const float* __restrict__ sal, int* __restrict__ idx) {
  __shared__ float sv[256];
  __shared__ float red[256];
  __shared__ int   redi[256];
  const int bt = blockIdx.x, tid = threadIdx.x;
  sv[tid] = (tid < 196) ? sal[bt * 196 + tid] : -1e30f;
  __syncthreads();
  for (int k = 0; k < 8; ++k) {
    red[tid] = sv[tid]; redi[tid] = tid;
    __syncthreads();
    for (int s = 128; s > 0; s >>= 1) {
      if (tid < s && red[tid + s] > red[tid]) { red[tid] = red[tid + s]; redi[tid] = redi[tid + s]; }
      __syncthreads();
    }
    if (tid == 0) { idx[bt * 8 + k] = redi[0]; sv[redi[0]] = -1e30f; }
    __syncthreads();
  }
}

// anchor gather: tok[(b*8+k)*16+t][a] = (ph+delta)[b,t,idx,a]
__global__ void gather_kernel(const float* __restrict__ ph, const float* __restrict__ delta,
                              const int* __restrict__ idx, float* __restrict__ tok) {
  long i = (long)blockIdx.x * 256 + threadIdx.x;
  if (i >= (long)2048 * 192) return;
  int a = (int)(i % 192);
  long r = i / 192;
  int t = (int)(r % 16);
  long bkq = r / 16;
  int k = (int)(bkq % 8);
  int b = (int)(bkq / 8);
  int n = idx[(b * 16 + t) * 8 + k];
  long src = (((long)(b * 16 + t)) * 196 + n) * 192 + a;
  tok[i] = ph[src] + delta[src];
}

// small attention: 128 sequences x 4 heads, T=16, dh=48
__global__ void attn_kernel(const float* __restrict__ qkv, float* __restrict__ out) {
  __shared__ float s[16][16];
  __shared__ float att[16][16];
  const int bk = blockIdx.x;
  const int t1 = threadIdx.x >> 4, t2 = threadIdx.x & 15;
  const float* base = qkv + (long)bk * 16 * 576;
  for (int h = 0; h < 4; ++h) {
    const float* qp = base + t1 * 576 + h * 48;
    const float* kp = base + t2 * 576 + 192 + h * 48;
    float acc = 0.0f;
    for (int d = 0; d < 48; ++d) acc += qp[d] * kp[d];
    s[t1][t2] = acc * 0.14433756729740646f;   // 1/sqrt(48)
    __syncthreads();
    if (t2 == 0) {
      float mx = -1e30f;
      for (int j = 0; j < 16; ++j) mx = fmaxf(mx, s[t1][j]);
      float sm = 0.0f;
      for (int j = 0; j < 16; ++j) { float e = __expf(s[t1][j] - mx); att[t1][j] = e; sm += e; }
      float inv = 1.0f / sm;
      for (int j = 0; j < 16; ++j) att[t1][j] *= inv;
    }
    __syncthreads();
    for (int it = 0; it < 3; ++it) {
      int e = threadIdx.x + it * 256;
      int r = e / 48, d = e % 48;
      float o = 0.0f;
      for (int j = 0; j < 16; ++j) o += att[r][j] * base[j * 576 + 384 + h * 48 + d];
      out[((long)bk * 16 + r) * 192 + h * 48 + d] = o;
    }
    __syncthreads();
  }
}

// scatter anchor tokens into anchor_map (pre-zeroed) + accumulate ap
__global__ void scatter_kernel(const float* __restrict__ aout, const int* __restrict__ idx,
                               float* __restrict__ amap, float* __restrict__ ap) {
  long i = (long)blockIdx.x * 256 + threadIdx.x;
  if (i >= (long)2048 * 192) return;
  int a = (int)(i % 192);
  long r = i / 192;
  int t = (int)(r % 16);
  long bkq = r / 16;
  int k = (int)(bkq % 8);
  int b = (int)(bkq / 8);
  int n = idx[(b * 16 + t) * 8 + k];
  float v = aout[i];
  amap[(((long)(b * 16 + t)) * 196 + n) * 192 + a] = v;  // top-k indices distinct
  atomicAdd(&ap[b * 192 + a], v * (1.0f / 128.0f));
}

// per-batch gate MLP + 3-way softmax
__global__ void gate_kernel(const float* __restrict__ dp, const float* __restrict__ ap,
                            const float* __restrict__ w1, const float* __restrict__ b1,
                            const float* __restrict__ w2, const float* __restrict__ b2,
                            float* __restrict__ gw) {
  __shared__ float gin[384];
  __shared__ float h[192];
  __shared__ float g2[576];
  const int b = blockIdx.x, tid = threadIdx.x;
  if (tid < 192)      gin[tid] = dp[b * 192 + tid];
  else if (tid < 384) gin[tid] = ap[b * 192 + tid - 192];
  __syncthreads();
  if (tid < 192) {
    float s = b1[tid];
    for (int k = 0; k < 384; ++k) s += gin[k] * w1[tid * 384 + k];
    h[tid] = gelu_f(s);
  }
  __syncthreads();
  for (int j = tid; j < 576; j += 256) {
    float s = b2[j];
    for (int k = 0; k < 192; ++k) s += h[k] * w2[j * 192 + k];
    g2[j] = s;
  }
  __syncthreads();
  if (tid < 192) {
    float m  = fmaxf(fmaxf(g2[tid], g2[192 + tid]), g2[384 + tid]);
    float e0 = __expf(g2[tid] - m);
    float e1 = __expf(g2[192 + tid] - m);
    float e2 = __expf(g2[384 + tid] - m);
    float inv = 1.0f / (e0 + e1 + e2);
    gw[(b * 3 + 0) * 192 + tid] = e0 * inv;
    gw[(b * 3 + 1) * 192 + tid] = e1 * inv;
    gw[(b * 3 + 2) * 192 + tid] = e2 * inv;
  }
}

// fused = gw0*local + gw1*trans + gw2*amap ; accumulate fmean over n
__global__ void fuse_kernel(const float* __restrict__ local_, const float* __restrict__ trans,
                            const float* __restrict__ amap, const float* __restrict__ gw,
                            float* __restrict__ fused, float* __restrict__ fmean) {
  long i = (long)blockIdx.x * 256 + threadIdx.x;
  if (i >= (long)50176 * 192) return;
  int a = (int)(i % 192);
  long r = i / 192;
  long btq = r / 196;
  int b = (int)(btq / 16);
  const float* g = gw + (long)b * 3 * 192;
  float f = g[a] * local_[i] + g[192 + a] * trans[i] + g[384 + a] * amap[i];
  fused[i] = f;
  atomicAdd(&fmean[btq * 192 + a], f * (1.0f / 196.0f));
}

// ---------------------------------------------------------------------------
extern "C" void kernel_launch(void* const* d_in, const int* in_sizes, int n_in,
                              void* d_out, int out_size, void* d_ws, size_t ws_size,
                              hipStream_t stream) {
  (void)in_sizes; (void)n_in; (void)out_size; (void)ws_size;
  constexpr int  Bb = 16, Tt = 16, Np = 196, Aa = 192, Cc = 768;
  constexpr long M1 = (long)Bb * Tt * Np;     // 50176
  constexpr long E1 = M1 * Aa;                // 9,633,792

  const float* x        = (const float*)d_in[0];
  const float* down_w   = (const float*)d_in[1];
  const float* down_b   = (const float*)d_in[2];
  const float* ldw_w    = (const float*)d_in[3];
  const float* lpw_w    = (const float*)d_in[4];
  const float* lnorm_g  = (const float*)d_in[5];
  const float* lnorm_b  = (const float*)d_in[6];
  const float* tdw_w    = (const float*)d_in[7];
  const float* tpw_w    = (const float*)d_in[8];
  const float* tmlp_w1  = (const float*)d_in[9];
  const float* tmlp_b1  = (const float*)d_in[10];
  const float* tmlp_w2  = (const float*)d_in[11];
  const float* tmlp_b2  = (const float*)d_in[12];
  const float* tnorm_g  = (const float*)d_in[13];
  const float* tnorm_b  = (const float*)d_in[14];
  const float* attnin_w = (const float*)d_in[15];
  const float* attnin_b = (const float*)d_in[16];
  const float* attnout_w= (const float*)d_in[17];
  const float* attnout_b= (const float*)d_in[18];
  const float* aproj_w  = (const float*)d_in[19];
  const float* aproj_b  = (const float*)d_in[20];
  const float* anorm_g  = (const float*)d_in[21];
  const float* anorm_b  = (const float*)d_in[22];
  const float* gate_w1  = (const float*)d_in[23];
  const float* gate_b1  = (const float*)d_in[24];
  const float* gate_w2  = (const float*)d_in[25];
  const float* gate_b2  = (const float*)d_in[26];
  const float* up_w     = (const float*)d_in[27];
  const float* up_b     = (const float*)d_in[28];
  const float* cls_w    = (const float*)d_in[29];
  const float* cls_b    = (const float*)d_in[30];
  float* out = (float*)d_out;

  // ---- workspace carve-up (floats) ----
  float* ws = (float*)d_ws;
  long off = 0;
  auto alloc = [&](long n) { float* p = ws + off; off += n; return p; };
  float* PH    = alloc(E1);
  float* DELTA = alloc(E1);
  float* LOCAL = alloc(E1);        // later reused as FUSED (in-place)
  float* TRANS = alloc(E1);
  float* AMAP  = alloc(E1);
  float* TMPA  = alloc(E1);
  float* TMPC  = alloc(E1);
  float* TMPB  = alloc(2 * E1);    // h1 (50176 x 384)
  float* SAL   = alloc(M1);
  int*   IDX   = (int*)alloc(4096);
  float* TOK   = alloc(2048L * 192);
  float* QKV   = alloc(2048L * 576);
  float* AT1   = alloc(2048L * 192);
  float* AT2   = alloc(2048L * 192);
  float* FMEAN = alloc(256L * 192);
  float* DP    = alloc(16L * 192);
  float* AP    = alloc(16L * 192);
  float* GW    = alloc(16L * 3 * 192);
  _Float16* wh = (_Float16*)(ws + off);
  long hoff = 0;
  auto halloc = [&](long n) { _Float16* p = wh + hoff; hoff += n; return p; };
  _Float16* Wdown = halloc((long)Aa * Cc);        // [192][768]
  _Float16* Wlpw  = halloc((long)Aa * Aa);
  _Float16* Wtpw  = halloc((long)Aa * Aa);
  _Float16* Wtm1  = halloc((long)2 * Aa * Aa);    // [384][192]
  _Float16* Wtm2  = halloc((long)Aa * 2 * Aa);    // [192][384]
  _Float16* Wain  = halloc((long)3 * Aa * Aa);    // [576][192]
  _Float16* Waout = halloc((long)Aa * Aa);
  _Float16* Wapr  = halloc((long)Aa * Aa);
  _Float16* Wup   = halloc((long)Cc * Aa);        // [768][192]
  _Float16* Wcls  = halloc((long)Cc * Aa);

  // ---- zero accumulators ----
  (void)hipMemsetAsync(AMAP,  0, E1 * sizeof(float), stream);
  (void)hipMemsetAsync(FMEAN, 0, 256L * 192 * sizeof(float), stream);
  (void)hipMemsetAsync(AP,    0, 16L * 192 * sizeof(float), stream);

  // ---- weight conversion fp32 -> f16 (layout preserved: [N][K]) ----
  auto wcv = [&](const float* W, _Float16* O, long n) {
    wcast_kernel<<<(unsigned)((n + 255) / 256), 256, 0, stream>>>(W, O, n);
  };
  wcv(down_w,    Wdown, (long)Aa * Cc);
  wcv(lpw_w,     Wlpw,  (long)Aa * Aa);
  wcv(tpw_w,     Wtpw,  (long)Aa * Aa);
  wcv(tmlp_w1,   Wtm1,  (long)2 * Aa * Aa);
  wcv(tmlp_w2,   Wtm2,  (long)Aa * 2 * Aa);
  wcv(attnin_w,  Wain,  (long)3 * Aa * Aa);
  wcv(attnout_w, Waout, (long)Aa * Aa);
  wcv(aproj_w,   Wapr,  (long)Aa * Aa);
  wcv(up_w,      Wup,   (long)Cc * Aa);
  wcv(cls_w,     Wcls,  (long)Cc * Aa);

  auto gemm = [&](const float* Ap, int lda, int ag, int apd, int ah,
                  const _Float16* Btp, const float* biasp,
                  const float* residp, int ldr,
                  float* Cp, int ldc, int cg, int cp, int ch,
                  int M, int Nc, int K, int mode) {
    dim3 grid(Nc / 64, M / 128);
    gemm_wmma_kernel<<<grid, 256, 0, stream>>>(Ap, lda, ag, apd, ah, Btp, K, biasp,
                                               residp, ldr, Cp, ldc, cg, cp, ch,
                                               M, Nc, K, mode);
  };
  const unsigned gE = (unsigned)((E1 + 255) / 256);

  // 1) ph = patch @ down_w^T + down_b   (A rows remapped past cls token)
  gemm(x, Cc, Np, 1, 1, Wdown, down_b, nullptr, 0, PH, Aa, (int)M1, 0, 0, (int)M1, Aa, Cc, 0);

  // 2) delta + saliency + per-batch delta pooling
  delta_sal_kernel<<<(unsigned)(M1 / 8), 256, 0, stream>>>(PH, DELTA, SAL);
  dp_kernel<<<12, 256, 0, stream>>>(DELTA, DP);

  // 3) local branch: dw3x3+gelu -> pointwise -> LN
  conv3x3_kernel<<<gE, 256, 0, stream>>>(PH, ldw_w, TMPA);
  gemm(TMPA, Aa, (int)M1, 0, 0, Wlpw, nullptr, nullptr, 0, TMPC, Aa, (int)M1, 0, 0, (int)M1, Aa, Aa, 0);
  ln_kernel<<<(unsigned)(M1 / 8), 256, 0, stream>>>(TMPC, LOCAL, lnorm_g, lnorm_b);

  // 4) temporal branch: dw-t conv+gelu -> pointwise -> MLP -> LN
  tconv_kernel<<<gE, 256, 0, stream>>>(DELTA, tdw_w, TMPA);
  gemm(TMPA, Aa, (int)M1, 0, 0, Wtpw, nullptr, nullptr, 0, TMPC, Aa, (int)M1, 0, 0, (int)M1, Aa, Aa, 0);
  gemm(TMPC, Aa, (int)M1, 0, 0, Wtm1, tmlp_b1, nullptr, 0, TMPB, 2 * Aa, (int)M1, 0, 0, (int)M1, 2 * Aa, Aa, 1);
  gemm(TMPB, 2 * Aa, (int)M1, 0, 0, Wtm2, tmlp_b2, nullptr, 0, TMPA, Aa, (int)M1, 0, 0, (int)M1, Aa, 2 * Aa, 0);
  ln_kernel<<<(unsigned)(M1 / 8), 256, 0, stream>>>(TMPA, TRANS, tnorm_g, tnorm_b);

  // 5) top-8 anchors + gather + attention
  topk_kernel<<<256, 256, 0, stream>>>(SAL, IDX);
  gather_kernel<<<(unsigned)((2048L * 192 + 255) / 256), 256, 0, stream>>>(PH, DELTA, IDX, TOK);
  gemm(TOK, Aa, 2048, 0, 0, Wain, attnin_b, nullptr, 0, QKV, 3 * Aa, 2048, 0, 0, 2048, 3 * Aa, Aa, 0);
  attn_kernel<<<128, 256, 0, stream>>>(QKV, AT1);
  gemm(AT1, Aa, 2048, 0, 0, Waout, attnout_b, nullptr, 0, AT2, Aa, 2048, 0, 0, 2048, Aa, Aa, 0);
  gemm(AT2, Aa, 2048, 0, 0, Wapr, aproj_b, nullptr, 0, AT1, Aa, 2048, 0, 0, 2048, Aa, Aa, 0);

  // 6) scatter into anchor map + LN (in-place, row-local) + anchor pooling
  scatter_kernel<<<(unsigned)((2048L * 192 + 255) / 256), 256, 0, stream>>>(AT1, IDX, AMAP, AP);
  ln_kernel<<<(unsigned)(M1 / 8), 256, 0, stream>>>(AMAP, AMAP, anorm_g, anorm_b);

  // 7) gate + fuse (fused written in-place over LOCAL)
  gate_kernel<<<16, 256, 0, stream>>>(DP, AP, gate_w1, gate_b1, gate_w2, gate_b2, GW);
  fuse_kernel<<<gE, 256, 0, stream>>>(LOCAL, TRANS, AMAP, GW, LOCAL, FMEAN);

  // 8) outputs: patch rows (residual from x, store remapped into d_out)
  gemm(LOCAL, Aa, (int)M1, 0, 0, Wup, up_b, x, Cc, out, Cc, Np, 1, 1, (int)M1, Cc, Aa, 2);
  // cls rows: r -> r*197
  gemm(FMEAN, Aa, 256, 0, 0, Wcls, cls_b, x, Cc, out, Cc, 1, Np, 0, 256, Cc, Aa, 2);
}
